// BiSpectrumLoss_20237885899536
// MI455X (gfx1250) — matrix operand order repaired and checked by reading, not compiled, not run on hardware
//
#include <hip/hip_runtime.h>

typedef _Float16 h8   __attribute__((ext_vector_type(8)));
typedef _Float16 v16h __attribute__((ext_vector_type(16)));
typedef float    v8f  __attribute__((ext_vector_type(8)));

#define N_FFT   512
#define HOP     160
#define NFEAT   257
#define NPAD    272          // 17 tiles of 16 bins
#define TFRAMES 601
#define LSIG    96000
#define BATCH   32
#define EPSV    1e-8f

// ---------------------------------------------------------------------------
// Init: build f16 DFT basis (Re = cos, Im = -sin) in workspace, zero d_out.
// Layout: basis[0 .. NPAD*512)        = Re[n][k]
//         basis[NPAD*512 .. 2*NPAD*512) = Im[n][k]
// ---------------------------------------------------------------------------
__global__ void bispec_init(_Float16* __restrict__ basis, float* __restrict__ out) {
    int idx = blockIdx.x * blockDim.x + threadIdx.x;
    if (idx == 0) out[0] = 0.0f;
    if (idx >= NPAD * N_FFT) return;
    int n = idx >> 9;       // / 512
    int k = idx & 511;
    float re = 0.0f, im = 0.0f;
    if (n < NFEAT) {
        int m = (n * k) & 511;                       // exact arg reduction
        float th = (float)m * (6.28318530717958647692f / 512.0f);
        re =  cosf(th);
        im = -sinf(th);
    }
    basis[idx]                  = (_Float16)re;
    basis[NPAD * N_FFT + idx]   = (_Float16)im;
}

// ---------------------------------------------------------------------------
// Fused STFT (WMMA GEMM) + bi-spectrum polar loss.
// Block: one 16-frame tile of one batch row, both signals.
// ---------------------------------------------------------------------------
__global__ void bispec_main(const float* __restrict__ preds,
                            const float* __restrict__ targets,
                            const float* __restrict__ window,
                            const int*   __restrict__ lengths,
                            const _Float16* __restrict__ basis,
                            float* __restrict__ out) {
    extern __shared__ unsigned char smem[];
    _Float16* Ald = (_Float16*)smem;                                   // [2][16][512] f16
    float*    Sld = (float*)(smem + 2 * 16 * N_FFT * sizeof(_Float16)); // [2][2][16][NPAD] f32
    float*    red = (float*)(smem + 2 * 16 * N_FFT * sizeof(_Float16)
                                  + 2 * 2 * 16 * NPAD * sizeof(float)); // [256]

    const int tid  = threadIdx.x;
    const int b    = blockIdx.y;
    const int tile = blockIdx.x;

    // ---- Phase A: windowed frames (reflect pad), f16, into LDS -------------
    const float* xs0 = preds   + (size_t)b * LSIG;
    const float* xs1 = targets + (size_t)b * LSIG;
    for (int i = tid; i < 2 * 16 * N_FFT; i += blockDim.x) {
        int s   = i >> 13;           // / 8192
        int rem = i & 8191;
        int row = rem >> 9;
        int k   = rem & 511;
        int j   = (tile * 16 + row) * HOP + k - (N_FFT / 2);
        if (j < 0)      j = -j;                 // reflect (no edge repeat)
        if (j >= LSIG)  j = 2 * LSIG - 2 - j;
        const float* x = s ? xs1 : xs0;
        Ald[i] = (_Float16)(x[j] * window[k]);
    }
    __syncthreads();

    // ---- Phase B: S[s][re/im][16 frames][NPAD bins] via v_wmma -------------
    const int lane   = tid & 31;
    const int wave   = tid >> 5;
    const int nwaves = blockDim.x >> 5;
    const int lm     = lane & 15;     // row of A / col of B
    const int hi     = lane >> 4;     // lane-half select

    for (int combo = wave; combo < 68; combo += nwaves) {
        int s = combo & 1;            // signal
        int p = (combo >> 1) & 1;     // 0 = Re, 1 = Im
        int j = combo >> 2;           // N-tile (0..16)
        const _Float16* Arow = Ald + (s * 16 + lm) * N_FFT;
        const _Float16* Brow = basis + (size_t)p * (NPAD * N_FFT)
                                     + (size_t)(j * 16 + lm) * N_FFT;
        v8f acc = {0.f, 0.f, 0.f, 0.f, 0.f, 0.f, 0.f, 0.f};
        for (int kk = 0; kk < 16; ++kk) {
            // A k order used here (contiguous, same as B): lanes 0-15 K=0..15, lanes 16-31 K=16..31
            int bk = kk * 32 + hi * 16;
            h8 alo = *(const h8*)(Arow + bk);
            h8 ahi = *(const h8*)(Arow + bk + 8);
            v16h av = __builtin_shufflevector(alo, ahi,
                       0, 1, 2, 3, 4, 5, 6, 7, 8, 9, 10, 11, 12, 13, 14, 15);
            // B k order used here (contiguous, same as A): lanes 0-15 K=0..15, lanes 16-31 K=16..31
            v16h bv = *(const v16h*)(Brow + kk * 32 + hi * 16);
            acc = __builtin_amdgcn_wmma_f32_16x16x32_f16(
                      false, av, false, bv, (short)0, acc, false, false);
        }
        // C/D layout: VGPR r -> M = r + 8*hi ; N = j*16 + lm
        float* Sout = Sld + ((s * 2 + p) * 16) * NPAD + (j * 16 + lm);
        #pragma unroll
        for (int r = 0; r < 8; ++r) {
            Sout[(r + hi * 8) * NPAD] = acc[r];
        }
    }
    __syncthreads();

    // ---- Phase C: bi-spectrum polar loss ----------------------------------
#define SRE(s, tl, f) Sld[(((s) * 2 + 0) * 16 + (tl)) * NPAD + (f)]
#define SIM(s, tl, f) Sld[(((s) * 2 + 1) * 16 + (tl)) * NPAD + (f)]
    const int nv = 1 + lengths[b] / HOP;    // valid frame count
    float local = 0.0f;
    for (int idx = tid; idx < 16 * NFEAT; idx += blockDim.x) {
        int tl = idx / NFEAT;
        int f1 = idx - tl * NFEAT;
        int t  = tile * 16 + tl;
        if (t >= TFRAMES || t >= nv) continue;
        int f2 = (f1 < (NFEAT - 1) - f1) ? f1 : (NFEAT - 1) - f1;
        int f3 = f1 + f2;

        // preds bi-spectrum
        float pr1 = SRE(0, tl, f1), pi1 = SIM(0, tl, f1);
        float pr2 = SRE(0, tl, f2), pi2 = SIM(0, tl, f2);
        float pr3 = SRE(0, tl, f3), pi3 = SIM(0, tl, f3);
        float txr = pr1 * pr2 - pi1 * pi2;
        float txi = pr1 * pi2 + pi1 * pr2;
        float bpr = txr * pr3 + txi * pi3;      // * conj(S3)
        float bpi = txi * pr3 - txr * pi3;

        // targets bi-spectrum
        float qr1 = SRE(1, tl, f1), qi1 = SIM(1, tl, f1);
        float qr2 = SRE(1, tl, f2), qi2 = SIM(1, tl, f2);
        float qr3 = SRE(1, tl, f3), qi3 = SIM(1, tl, f3);
        float tyr = qr1 * qr2 - qi1 * qi2;
        float tyi = qr1 * qi2 + qi1 * qr2;
        float btr = tyr * qr3 + tyi * qi3;
        float bti = tyi * qr3 - tyr * qi3;

        float lp = logf(sqrtf(bpr * bpr + bpi * bpi) + EPSV);
        float lt = logf(sqrtf(btr * btr + bti * bti) + EPSV);
        float ap = atan2f(bpi, bpr);
        float at = atan2f(bti, btr);
        local += fabsf(lp - lt) + fabsf(ap - at);
    }
#undef SRE
#undef SIM

    // ---- block reduction + scalar accumulate ------------------------------
    red[tid] = local;
    __syncthreads();
    for (int s2 = blockDim.x >> 1; s2 > 0; s2 >>= 1) {
        if (tid < s2) red[tid] += red[tid + s2];
        __syncthreads();
    }
    if (tid == 0) atomicAdd(out, red[0]);
}

// ---------------------------------------------------------------------------
extern "C" void kernel_launch(void* const* d_in, const int* in_sizes, int n_in,
                              void* d_out, int out_size, void* d_ws, size_t ws_size,
                              hipStream_t stream) {
    (void)in_sizes; (void)n_in; (void)out_size; (void)ws_size;

    const float* preds   = (const float*)d_in[0];
    const float* targets = (const float*)d_in[1];
    const float* window  = (const float*)d_in[2];
    const int*   lengths = (const int*)d_in[3];
    float*       out     = (float*)d_out;
    _Float16*    basis   = (_Float16*)d_ws;   // needs 2*NPAD*512*2 = 557 056 B

    const int initThreads = 256;
    const int initBlocks  = (NPAD * N_FFT + initThreads - 1) / initThreads;
    bispec_init<<<initBlocks, initThreads, 0, stream>>>(basis, out);

    const size_t shmem = 2 * 16 * N_FFT * sizeof(_Float16)   // A tiles   32 768 B
                       + 2 * 2 * 16 * NPAD * sizeof(float)   // S tiles   69 632 B
                       + 256 * sizeof(float);                // reduction  1 024 B
    // Allow >64KB dynamic LDS where the runtime wants opt-in (harmless if not).
    (void)hipFuncSetAttribute((const void*)bispec_main,
                              hipFuncAttributeMaxDynamicSharedMemorySize,
                              (int)shmem);

    const int TILES = (TFRAMES + 15) / 16;   // 38
    bispec_main<<<dim3(TILES, BATCH), 256, shmem, stream>>>(
        preds, targets, window, lengths, basis, out);
}
